// TransformerBlock_41850161332444
// MI455X (gfx1250) — matrix operand order repaired and checked by reading, not compiled, not run on hardware
//
#include <hip/hip_runtime.h>
#include <hip/hip_bf16.h>
#include <math.h>

#define B_   4
#define S_   1024
#define D_   768
#define H_   12
#define HD_  64
#define DFF_ 3072
#define M_   (B_ * S_)   // 4096 rows

typedef __bf16 bf16;
typedef __attribute__((ext_vector_type(16))) __bf16 v16bf;
typedef __attribute__((ext_vector_type(8)))  float  v8f;

// D = A(16x32 bf16) * B(32x16 bf16) + C(16x16 f32), wave32
__device__ __forceinline__ v8f wmma_bf16(v16bf a, v16bf b, v8f c) {
  return __builtin_amdgcn_wmma_f32_16x16x32_bf16(
      /*neg_a=*/false, a, /*neg_b=*/false, b,
      /*c_mod=*/(short)0, c, /*reuse_a=*/false, /*reuse_b=*/false);
}

__device__ __forceinline__ v8f zero_v8f() {
  v8f z;
#pragma unroll
  for (int i = 0; i < 8; ++i) z[i] = 0.0f;
  return z;
}

// ---------------------------------------------------------------------------
// Fragment-swizzled storage layouts (one 16x16x32 tile = 512 elements, stored
// as lane*16 + e so each lane reads 32 contiguous bytes):
//   A-frag (16x32, MxK):  lane = (m&15) + 16*((k>>3)&1), e = ((k>>4)<<3)+(k&7)
//   B-frag (32x16, KxN):  lane = (n&15) + 16*((k>>3)&1), e = ((k>>4)<<3)+(k&7)
// Tiles ordered: (m-tile or n-tile) major, k-tile inner => k-advance is +1KB.
// ---------------------------------------------------------------------------

// fp32 W[K,N] -> bf16 B-fragment-swizzled
__global__ void cvt_w_bswz(const float* __restrict__ W, bf16* __restrict__ out,
                           int Kd, int Nd) {
  int i = blockIdx.x * 256 + threadIdx.x;
  int k = i / Nd;
  int n = i - k * Nd;
  float v = W[i];
  int kk = k & 31;
  size_t tile = (size_t)(n >> 4) * (Kd >> 5) + (k >> 5);
  int lp = (n & 15) + (((kk >> 3) & 1) << 4);
  int e  = ((kk >> 4) << 3) + (kk & 7);
  out[tile * 512 + lp * 16 + e] = (bf16)v;
}

// LayerNorm over rows of 768, output bf16 in A-fragment-swizzled layout
__global__ void ln_aswz(const float* __restrict__ x, const float* __restrict__ gam,
                        const float* __restrict__ bet, bf16* __restrict__ out) {
  __shared__ float red[256];
  const int row = blockIdx.x;
  const int tid = threadIdx.x;
  const float* xr = x + (size_t)row * D_;
  float v[3];
  float s = 0.0f;
#pragma unroll
  for (int i = 0; i < 3; ++i) { v[i] = xr[tid + i * 256]; s += v[i]; }
  red[tid] = s; __syncthreads();
  for (int off = 128; off > 0; off >>= 1) {
    if (tid < off) red[tid] += red[tid + off];
    __syncthreads();
  }
  float mu = red[0] * (1.0f / 768.0f);
  __syncthreads();
  float s2 = 0.0f;
#pragma unroll
  for (int i = 0; i < 3; ++i) { float d = v[i] - mu; s2 += d * d; }
  red[tid] = s2; __syncthreads();
  for (int off = 128; off > 0; off >>= 1) {
    if (tid < off) red[tid] += red[tid + off];
    __syncthreads();
  }
  float var = red[0] * (1.0f / 768.0f);
  float rs = rsqrtf(var + 1e-6f);
#pragma unroll
  for (int i = 0; i < 3; ++i) {
    int c = tid + i * 256;
    float y = (v[i] - mu) * rs * gam[c] + bet[c];
    int kk = c & 31;
    size_t tile = (size_t)(row >> 4) * 24 + (c >> 5);   // 768/32 = 24 k-tiles
    int lp = (row & 15) + (((kk >> 3) & 1) << 4);
    int e  = ((kk >> 4) << 3) + (kk & 7);
    out[tile * 512 + lp * 16 + e] = (bf16)y;
  }
}

#define EPI_Q    0
#define EPI_K    1
#define EPI_V    2
#define EPI_RES  3
#define EPI_GELU 4

// C[M,N] = Aswz[M,K] @ Wswz[K,N] + bias
// Wave tile 32x32: 4 accumulators, double-buffered fragments (SW pipeline),
// peeled tail so the steady-state loop has no clamp/select address math.
__global__ void gemm_bf16_wmma(const bf16* __restrict__ A, const bf16* __restrict__ W,
                               const float* __restrict__ bias, int Kd, int Nd, int mode,
                               const float* __restrict__ residual,
                               float* __restrict__ outF, bf16* __restrict__ outB) {
  const int lane = threadIdx.x & 31;
  const int wave = (blockIdx.x * blockDim.x + threadIdx.x) >> 5;
  const int nStrips = Nd >> 5;       // 32-col strips
  const int mt = wave / nStrips;     // 32-row tile
  const int ns = wave % nStrips;     // 32-col strip
  const int kTiles = Kd >> 5;        // 24 or 96 (even, >= 4)

  const v16bf* Av = reinterpret_cast<const v16bf*>(A);
  const v16bf* Wv = reinterpret_cast<const v16bf*>(W);

  // per-wave base pointers; k-tile stride = 32 v16bf chunks = 1 KB
  const v16bf* aP0 = Av + (size_t)(mt * 2 + 0) * kTiles * 32 + lane;
  const v16bf* aP1 = Av + (size_t)(mt * 2 + 1) * kTiles * 32 + lane;
  const v16bf* bP0 = Wv + (size_t)(ns * 2 + 0) * kTiles * 32 + lane;
  const v16bf* bP1 = Wv + (size_t)(ns * 2 + 1) * kTiles * 32 + lane;

  v8f acc[2][2];
#pragma unroll
  for (int mi = 0; mi < 2; ++mi)
#pragma unroll
    for (int nt = 0; nt < 2; ++nt) acc[mi][nt] = zero_v8f();

  // pipeline buffers (even/odd k-tiles)
  v16bf a0A = aP0[0], a0B = aP1[0], b0A = bP0[0], b0B = bP1[0];

  int kt = 0;
  for (; kt + 2 < kTiles; kt += 2) {
    const int o1 = (kt + 1) * 32;
    v16bf a1A = aP0[o1], a1B = aP1[o1], b1A = bP0[o1], b1B = bP1[o1];

    acc[0][0] = wmma_bf16(a0A, b0A, acc[0][0]);
    acc[1][0] = wmma_bf16(a0B, b0A, acc[1][0]);
    acc[0][1] = wmma_bf16(a0A, b0B, acc[0][1]);
    acc[1][1] = wmma_bf16(a0B, b0B, acc[1][1]);

    const int o2 = (kt + 2) * 32;
    a0A = aP0[o2]; a0B = aP1[o2]; b0A = bP0[o2]; b0B = bP1[o2];

    acc[0][0] = wmma_bf16(a1A, b1A, acc[0][0]);
    acc[1][0] = wmma_bf16(a1B, b1A, acc[1][0]);
    acc[0][1] = wmma_bf16(a1A, b1B, acc[0][1]);
    acc[1][1] = wmma_bf16(a1B, b1B, acc[1][1]);
  }
  // tail: kt == kTiles-2, both tiles already addressable, no prefetch beyond end
  {
    const int o1 = (kt + 1) * 32;
    v16bf a1A = aP0[o1], a1B = aP1[o1], b1A = bP0[o1], b1B = bP1[o1];

    acc[0][0] = wmma_bf16(a0A, b0A, acc[0][0]);
    acc[1][0] = wmma_bf16(a0B, b0A, acc[1][0]);
    acc[0][1] = wmma_bf16(a0A, b0B, acc[0][1]);
    acc[1][1] = wmma_bf16(a0B, b0B, acc[1][1]);

    acc[0][0] = wmma_bf16(a1A, b1A, acc[0][0]);
    acc[1][0] = wmma_bf16(a1B, b1A, acc[1][0]);
    acc[0][1] = wmma_bf16(a1A, b1B, acc[0][1]);
    acc[1][1] = wmma_bf16(a1B, b1B, acc[1][1]);
  }

  const int g = lane >> 4;
#pragma unroll
  for (int mi = 0; mi < 2; ++mi) {
#pragma unroll
    for (int nt = 0; nt < 2; ++nt) {
#pragma unroll
      for (int j = 0; j < 8; ++j) {
        int m = (mt << 5) + (mi << 4) + j + (g << 3);
        int n = ((ns * 2 + nt) << 4) + (lane & 15);
        float val = acc[mi][nt][j] + bias[n];
        if (mode == EPI_RES) {
          size_t o = (size_t)m * Nd + n;
          outF[o] = val + residual[o];
        } else if (mode == EPI_GELU) {
          float gv = 0.5f * val * (1.0f + erff(val * 0.70710678118f));
          int kk = n & 31;
          size_t tile = (size_t)(m >> 4) * (Nd >> 5) + (n >> 5);
          int lp = (m & 15) + (((kk >> 3) & 1) << 4);
          int e  = ((kk >> 4) << 3) + (kk & 7);
          outB[tile * 512 + lp * 16 + e] = (bf16)gv;
        } else {
          // Q/K/V scatter into per-(b,h) fragment layouts (128 tiles each)
          int bb = m >> 10, t = m & 1023;
          int hh = n >> 6,  d = n & 63;
          size_t base = (size_t)(bb * H_ + hh) * 65536;
          if (mode == EPI_Q) {              // A-frag of Q (M'=1024, K'=64), pre-scaled
            float qv = val * 0.125f;        // 1/sqrt(64)
            int kk = d & 31;
            int tile = ((t >> 4) << 1) + (d >> 5);
            int lp = (t & 15) + (((kk >> 3) & 1) << 4);
            int e  = ((kk >> 4) << 3) + (kk & 7);
            outB[base + (size_t)tile * 512 + lp * 16 + e] = (bf16)qv;
          } else if (mode == EPI_K) {       // B-frag of K^T (k-dim = d, n = token)
            int kk = d & 31;
            int tile = ((t >> 4) << 1) + (d >> 5);
            int lp = (t & 15) + (((kk >> 3) & 1) << 4);
            int e  = ((kk >> 4) << 3) + (kk & 7);
            outB[base + (size_t)tile * 512 + lp * 16 + e] = (bf16)val;
          } else {                          // EPI_V: B-frag of V (k-dim = token, n = d)
            int tt = t & 31;
            int tile = ((d >> 4) << 5) + (t >> 5);
            int lp = (d & 15) + (((tt >> 3) & 1) << 4);
            int e  = ((tt >> 4) << 3) + (tt & 7);
            outB[base + (size_t)tile * 512 + lp * 16 + e] = (bf16)val;
          }
        }
      }
    }
  }
}

// Flash attention: one wave = 16 queries x 1024 keys, online softmax.
// K fragments double-buffered; V fragments issued before the softmax VALU
// section so their latency hides under it.
__global__ void attn_wmma(const bf16* __restrict__ Q, const bf16* __restrict__ K,
                          const bf16* __restrict__ V, bf16* __restrict__ ctxswz) {
  __shared__ __align__(32) bf16 pbuf[4][512];
  const int lane = threadIdx.x & 31;
  const int w = threadIdx.x >> 5;
  const int wave = blockIdx.x * 4 + w;
  const int qt = wave & 63;          // query tile of 16
  const int bh = wave >> 6;          // b*12 + h
  const int h = bh % H_;
  const int b = bh / H_;

  const v16bf* Qv = reinterpret_cast<const v16bf*>(Q) + (size_t)bh * 4096;
  const v16bf* Kv = reinterpret_cast<const v16bf*>(K) + (size_t)bh * 4096;
  const v16bf* Vv = reinterpret_cast<const v16bf*>(V) + (size_t)bh * 4096;

  v16bf qa0 = Qv[(qt * 2 + 0) * 32 + lane];   // d = 0..31
  v16bf qa1 = Qv[(qt * 2 + 1) * 32 + lane];   // d = 32..63

  v8f ctx[4] = {zero_v8f(), zero_v8f(), zero_v8f(), zero_v8f()};
  float m_i[8], l_i[8];
#pragma unroll
  for (int j = 0; j < 8; ++j) { m_i[j] = -3.0e38f; l_i[j] = 0.0f; }

  const int g = lane >> 4;
  const int c0 = lane & 15;
  const int gbit = (c0 >> 3) & 1;

  // preload K fragments for kb = 0
  v16bf kc0 = Kv[0 * 32 + lane];
  v16bf kc1 = Kv[1 * 32 + lane];
  v16bf kc2 = Kv[2 * 32 + lane];
  v16bf kc3 = Kv[3 * 32 + lane];

  for (int kb = 0; kb < 32; ++kb) {           // 32 keys per block
    v8f s0 = zero_v8f(), s1 = zero_v8f();
    s0 = wmma_bf16(qa0, kc0, s0);
    s0 = wmma_bf16(qa1, kc1, s0);
    s1 = wmma_bf16(qa0, kc2, s1);
    s1 = wmma_bf16(qa1, kc3, s1);

    // issue next-K + current-V loads; latency hides under softmax VALU work
    const int kbn = (kb + 1 < 32) ? kb + 1 : kb;
    v16bf kn0 = Kv[((2 * kbn + 0) * 2 + 0) * 32 + lane];
    v16bf kn1 = Kv[((2 * kbn + 0) * 2 + 1) * 32 + lane];
    v16bf kn2 = Kv[((2 * kbn + 1) * 2 + 0) * 32 + lane];
    v16bf kn3 = Kv[((2 * kbn + 1) * 2 + 1) * 32 + lane];
    v16bf vb0 = Vv[(0 * 32 + kb) * 32 + lane];
    v16bf vb1 = Vv[(1 * 32 + kb) * 32 + lane];
    v16bf vb2 = Vv[(2 * 32 + kb) * 32 + lane];
    v16bf vb3 = Vv[(3 * 32 + kb) * 32 + lane];

    float scl[8];
#pragma unroll
    for (int j = 0; j < 8; ++j) {
      float v0 = s0[j], v1 = s1[j];
      float mx = fmaxf(v0, v1);
      mx = fmaxf(mx, __shfl_xor(mx, 1, 32));
      mx = fmaxf(mx, __shfl_xor(mx, 2, 32));
      mx = fmaxf(mx, __shfl_xor(mx, 4, 32));
      mx = fmaxf(mx, __shfl_xor(mx, 8, 32));
      float mnew = fmaxf(m_i[j], mx);
      float sc = __expf(m_i[j] - mnew);
      float p0 = __expf(v0 - mnew);
      float p1 = __expf(v1 - mnew);
      float rs = p0 + p1;
      rs += __shfl_xor(rs, 1, 32);
      rs += __shfl_xor(rs, 2, 32);
      rs += __shfl_xor(rs, 4, 32);
      rs += __shfl_xor(rs, 8, 32);
      l_i[j] = l_i[j] * sc + rs;
      m_i[j] = mnew;
      scl[j] = sc;
      // stage P into per-wave LDS tile already in A-frag layout
      int row = j + (g << 3);
      pbuf[w][(row + (gbit << 4)) * 16 + (c0 & 7)]     = (bf16)p0;  // key c0
      pbuf[w][(row + (gbit << 4)) * 16 + 8 + (c0 & 7)] = (bf16)p1;  // key 16+c0
    }
#pragma unroll
    for (int dt = 0; dt < 4; ++dt)
#pragma unroll
      for (int j = 0; j < 8; ++j) ctx[dt][j] *= scl[j];

    asm volatile("s_wait_dscnt 0" ::: "memory");   // CDNA5 split DS counter
    v16bf pa = reinterpret_cast<const v16bf*>(pbuf[w])[lane];
    ctx[0] = wmma_bf16(pa, vb0, ctx[0]);
    ctx[1] = wmma_bf16(pa, vb1, ctx[1]);
    ctx[2] = wmma_bf16(pa, vb2, ctx[2]);
    ctx[3] = wmma_bf16(pa, vb3, ctx[3]);

    kc0 = kn0; kc1 = kn1; kc2 = kn2; kc3 = kn3;
  }

  // normalize and scatter context to A-frag layout for the Wo GEMM
#pragma unroll
  for (int j = 0; j < 8; ++j) {
    float inv = 1.0f / l_i[j];
    int rowg = b * S_ + (qt << 4) + j + (g << 3);
#pragma unroll
    for (int dt = 0; dt < 4; ++dt) {
      int d = (dt << 4) + (lane & 15);
      int k = h * 64 + d;
      float val = ctx[dt][j] * inv;
      int kk = k & 31;
      size_t tile = (size_t)(rowg >> 4) * 24 + (k >> 5);
      int lp = (rowg & 15) + (((kk >> 3) & 1) << 4);
      int e  = ((kk >> 4) << 3) + (kk & 7);
      ctxswz[tile * 512 + lp * 16 + e] = (bf16)val;
    }
  }
}

extern "C" void kernel_launch(void* const* d_in, const int* in_sizes, int n_in,
                              void* d_out, int out_size, void* d_ws, size_t ws_size,
                              hipStream_t stream) {
  (void)in_sizes; (void)n_in; (void)out_size; (void)ws_size;
  const float* x    = (const float*)d_in[0];
  const float* ln1g = (const float*)d_in[1];
  const float* ln1b = (const float*)d_in[2];
  const float* Wq   = (const float*)d_in[3];
  const float* bq   = (const float*)d_in[4];
  const float* Wk   = (const float*)d_in[5];
  const float* bk   = (const float*)d_in[6];
  const float* Wv   = (const float*)d_in[7];
  const float* bv   = (const float*)d_in[8];
  const float* Wo   = (const float*)d_in[9];
  const float* bo   = (const float*)d_in[10];
  const float* ln2g = (const float*)d_in[11];
  const float* ln2b = (const float*)d_in[12];
  const float* W1   = (const float*)d_in[13];
  const float* b1   = (const float*)d_in[14];
  const float* W2   = (const float*)d_in[15];
  const float* b2   = (const float*)d_in[16];
  float* out = (float*)d_out;

  char* ws = (char*)d_ws;
  size_t off = 0;
  auto take = [&](size_t bytes) -> void* {
    void* p = ws + off;
    off += (bytes + 255) & ~(size_t)255;
    return p;
  };
  bf16* wq_s  = (bf16*)take((size_t)D_ * D_ * 2);
  bf16* wk_s  = (bf16*)take((size_t)D_ * D_ * 2);
  bf16* wv_s  = (bf16*)take((size_t)D_ * D_ * 2);
  bf16* wo_s  = (bf16*)take((size_t)D_ * D_ * 2);
  bf16* w1_s  = (bf16*)take((size_t)D_ * DFF_ * 2);
  bf16* w2_s  = (bf16*)take((size_t)DFF_ * D_ * 2);
  bf16* xn1_s = (bf16*)take((size_t)M_ * D_ * 2);
  bf16* q_s   = (bf16*)take((size_t)M_ * D_ * 2);
  bf16* k_s   = (bf16*)take((size_t)M_ * D_ * 2);
  bf16* v_s   = (bf16*)take((size_t)M_ * D_ * 2);
  bf16* ctx_s = (bf16*)take((size_t)M_ * D_ * 2);
  float* x1   = (float*)take((size_t)M_ * D_ * 4);
  bf16* xn2_s = (bf16*)take((size_t)M_ * D_ * 2);
  bf16* h1_s  = (bf16*)take((size_t)M_ * DFF_ * 2);

  // 1) convert weights fp32 -> bf16 B-fragment-swizzled
  cvt_w_bswz<<<(D_ * D_) / 256, 256, 0, stream>>>(Wq, wq_s, D_, D_);
  cvt_w_bswz<<<(D_ * D_) / 256, 256, 0, stream>>>(Wk, wk_s, D_, D_);
  cvt_w_bswz<<<(D_ * D_) / 256, 256, 0, stream>>>(Wv, wv_s, D_, D_);
  cvt_w_bswz<<<(D_ * D_) / 256, 256, 0, stream>>>(Wo, wo_s, D_, D_);
  cvt_w_bswz<<<(D_ * DFF_) / 256, 256, 0, stream>>>(W1, w1_s, D_, DFF_);
  cvt_w_bswz<<<(DFF_ * D_) / 256, 256, 0, stream>>>(W2, w2_s, DFF_, D_);

  // 2) LN1 -> A-frag bf16
  ln_aswz<<<M_, 256, 0, stream>>>(x, ln1g, ln1b, xn1_s);

  // 3) Q/K/V projections (WMMA), scattered into attention fragment layouts
  const int gdD = (M_ / 32) * (D_ / 32) / 4;      // 768 blocks of 4 waves
  gemm_bf16_wmma<<<gdD, 128, 0, stream>>>(xn1_s, wq_s, bq, D_, D_, EPI_Q, nullptr, nullptr, q_s);
  gemm_bf16_wmma<<<gdD, 128, 0, stream>>>(xn1_s, wk_s, bk, D_, D_, EPI_K, nullptr, nullptr, k_s);
  gemm_bf16_wmma<<<gdD, 128, 0, stream>>>(xn1_s, wv_s, bv, D_, D_, EPI_V, nullptr, nullptr, v_s);

  // 4) flash attention (WMMA)
  attn_wmma<<<(B_ * H_ * (S_ / 16)) / 4, 128, 0, stream>>>(q_s, k_s, v_s, ctx_s);

  // 5) output projection + residual
  gemm_bf16_wmma<<<gdD, 128, 0, stream>>>(ctx_s, wo_s, bo, D_, D_, EPI_RES, x, x1, nullptr);

  // 6) LN2 -> A-frag bf16
  ln_aswz<<<M_, 256, 0, stream>>>(x1, ln2g, ln2b, xn2_s);

  // 7) MLP: GELU(xn2 @ W1 + b1) -> A-frag bf16
  const int gdF = (M_ / 32) * (DFF_ / 32) / 4;    // 3072 blocks
  gemm_bf16_wmma<<<gdF, 128, 0, stream>>>(xn2_s, w1_s, b1, D_, DFF_, EPI_GELU, nullptr, nullptr, h1_s);

  // 8) h1 @ W2 + b2 + residual -> out
  gemm_bf16_wmma<<<gdD, 128, 0, stream>>>(h1_s, w2_s, b2, DFF_, D_, EPI_RES, x1, out, nullptr);
}
